// DisentangledTransformer_2465311228230
// MI455X (gfx1250) — matrix operand order repaired and checked by reading, not compiled, not run on hardware
//
#include <hip/hip_runtime.h>
#include <hip/hip_bf16.h>

// DeBERTa-style disentangled attention, window half-width 8.
// B=16, N=512, D=512, H=8, d=64. Only |n-m|<=8 scores survive -> banded attention,
// only pos rows 504..520 are ever gathered.
//
// Stage 0: one-shot f32->f16 conversion of X / W / rel-window (removes cvt from hot loop).
// Stage 1: QKV + pos-window projections as f16 WMMA GEMMs, 32x64 tile per wave.
// Stage 2: banded attention per (b,h,16-query tile): QK^T / c2p / p2c / PV all WMMA.

typedef __attribute__((ext_vector_type(16))) _Float16 v16h;
typedef __attribute__((ext_vector_type(8)))  _Float16 v8h;
typedef __attribute__((ext_vector_type(8)))  float    v8f;

#define DMODEL 512
#define RSCALE 0.07216878364870323f   // 1/sqrt(64*3)

// ---- fragment builders -----------------------------------------------------
// A-matrix 16x32 f16 layout: lane L (M=L&15), kb=(L>>4)*8;
//   halves 0..7  -> K = kb + 0..7
//   halves 8..15 -> K = 16 + kb + 0..7
// B-matrix 32x16 f16 layout: lane L (N=L&15), same K order as A (kb=(L>>4)*8)
// C/D f32: lane L (N=L&15), vgpr r -> M = r + (L>>4)*8

__device__ __forceinline__ v16h mk16h(const _Float16* lo, const _Float16* hi) {
  v8h a = *(const v8h*)lo;
  v8h b = *(const v8h*)hi;
  v16h r;
#pragma unroll
  for (int i = 0; i < 8; ++i) { r[i] = a[i]; r[i + 8] = b[i]; }
  return r;
}

__device__ __forceinline__ v8f wmma32(v16h a, v16h b, v8f c) {
  return __builtin_amdgcn_wmma_f32_16x16x32_f16(false, a, false, b, (short)0, c,
                                                false, false);
}

// ---- Stage 0: bulk f32 -> f16 (n divisible by 8) ---------------------------
__global__ __launch_bounds__(256) void cvt_f16_kernel(
    const float* __restrict__ src, _Float16* __restrict__ dst, int n) {
  const int i = (blockIdx.x * 256 + threadIdx.x) * 8;
  if (i < n) {
    v8f x = *(const v8f*)(src + i);
    v8h y;
#pragma unroll
    for (int j = 0; j < 8; ++j) y[j] = (_Float16)x[j];
    *(v8h*)(dst + i) = y;
  }
}

// ---- Stage 1: Out[m,n] = sum_k X[m,k]*W[n,k] + bias[n], f16 in, f16 out ----
// grid.x = M/32 macro-tiles, grid.y = 8 (groups of 64 cols), block = 1 wave.
// Each wave: 2 A-tiles x 4 B-subtiles = 8 independent accumulator chains.
__global__ __launch_bounds__(32) void gemm_xwT_kernel(
    const _Float16* __restrict__ X, const _Float16* __restrict__ W,
    const float* __restrict__ bias, _Float16* __restrict__ Out) {
  const int lane = threadIdx.x & 31;
  const int l15  = lane & 15;
  const int hi   = lane >> 4;
  const int kbA  = hi * 8;
  const int kbB  = hi * 16;
  const int mt   = blockIdx.x;
  const int ng   = blockIdx.y;

  const _Float16* xrow0 = X + (mt * 32 + l15) * DMODEL;
  const _Float16* xrow1 = xrow0 + 16 * DMODEL;

  v8f acc[2][4] = {};

  for (int k0 = 0; k0 < DMODEL; k0 += 32) {
    __builtin_prefetch(xrow0 + k0 + 96, 0, 0);          // global_prefetch_b8
    __builtin_prefetch(xrow1 + k0 + 96, 0, 0);
    v16h a0 = mk16h(xrow0 + k0 + kbA, xrow0 + k0 + 16 + kbA);
    v16h a1 = mk16h(xrow1 + k0 + kbA, xrow1 + k0 + 16 + kbA);
#pragma unroll
    for (int nt = 0; nt < 4; ++nt) {
      const _Float16* wrow = W + (ng * 64 + nt * 16 + l15) * DMODEL;
      v16h b = mk16h(wrow + k0 + kbA, wrow + k0 + 16 + kbA);
      acc[0][nt] = wmma32(a0, b, acc[0][nt]);
      acc[1][nt] = wmma32(a1, b, acc[1][nt]);
    }
  }
#pragma unroll
  for (int half = 0; half < 2; ++half) {
#pragma unroll
    for (int nt = 0; nt < 4; ++nt) {
      const int ncol = ng * 64 + nt * 16 + l15;
      const float bv = bias[ncol];
#pragma unroll
      for (int r = 0; r < 8; ++r) {
        const int m = mt * 32 + half * 16 + r + hi * 8;
        Out[m * DMODEL + ncol] = (_Float16)(acc[half][nt][r] + bv);
      }
    }
  }
}

// ---- Stage 2: banded attention ---------------------------------------------
// grid = (32 q-tiles, H=8, B=16); block = 1 wave.
__global__ __launch_bounds__(32) void attn_kernel(
    const _Float16* __restrict__ qlh, const _Float16* __restrict__ klh,
    const _Float16* __restrict__ vlh, const _Float16* __restrict__ poskw,
    const _Float16* __restrict__ posqw, float* __restrict__ out) {
  const int lane = threadIdx.x & 31;
  const int l15  = lane & 15;
  const int hi   = lane >> 4;
  const int kbA  = hi * 8;
  const int kbB  = hi * 16;

  const int t  = blockIdx.x;      // query tile
  const int h  = blockIdx.y;
  const int b  = blockIdx.z;
  const int n0 = t * 16;
  const int m0 = n0 - 16;         // key window start (48 keys: tiles t-1,t,t+1)
  const int cbase = h * 64;

  __shared__ __align__(16) float    S[16][48];    // qk scores -> p
  __shared__ __align__(16) float    P[16][32];    // c2p table: P[n][j]
  __shared__ __align__(16) float    R[48][32];    // p2c table: R[m][j]
  __shared__ __align__(16) _Float16 VT[64][80];   // V^T: VT[dcol][kloc]
  __shared__ __align__(16) _Float16 PR[16][64];   // probs (f16, cols 48..63 = 0)

  // Q A-fragments (d=64 -> two K=32 steps), reused for QK^T and c2p.
  const _Float16* qrow = qlh + (b * 512 + n0 + l15) * DMODEL + cbase;
  v16h aq0 = mk16h(qrow + kbA,      qrow + 16 + kbA);
  v16h aq1 = mk16h(qrow + 32 + kbA, qrow + 48 + kbA);

  // ---- QK^T over 3 key tiles ----
#pragma unroll
  for (int kt = 0; kt < 3; ++kt) {
    const int key = m0 + kt * 16 + l15;
    v16h b0 = {}, b1 = {};
    if (key >= 0 && key < 512) {
      const _Float16* krow = klh + (b * 512 + key) * DMODEL + cbase;
      b0 = mk16h(krow + kbA,      krow + 16 + kbA);
      b1 = mk16h(krow + 32 + kbA, krow + 48 + kbA);
    }
    v8f acc = {};
    acc = wmma32(aq0, b0, acc);
    acc = wmma32(aq1, b1, acc);
#pragma unroll
    for (int r = 0; r < 8; ++r) S[r + hi * 8][kt * 16 + l15] = acc[r];
  }

  // ---- c2p: P[n][j] = ql[n] . pos_k[j], j = 0..16 (pad to 32) ----
#pragma unroll
  for (int jt = 0; jt < 2; ++jt) {
    const _Float16* prow = poskw + (jt * 16 + l15) * DMODEL + cbase;
    v16h b0 = mk16h(prow + kbA,      prow + 16 + kbA);
    v16h b1 = mk16h(prow + 32 + kbA, prow + 48 + kbA);
    v8f acc = {};
    acc = wmma32(aq0, b0, acc);
    acc = wmma32(aq1, b1, acc);
#pragma unroll
    for (int r = 0; r < 8; ++r) P[r + hi * 8][jt * 16 + l15] = acc[r];
  }

  // ---- p2c: R[m][j] = kl[m] . pos_q[j] ----
  v16h bq0[2], bq1[2];
#pragma unroll
  for (int jt = 0; jt < 2; ++jt) {
    const _Float16* prow = posqw + (jt * 16 + l15) * DMODEL + cbase;
    bq0[jt] = mk16h(prow + kbA,      prow + 16 + kbA);
    bq1[jt] = mk16h(prow + 32 + kbA, prow + 48 + kbA);
  }
#pragma unroll
  for (int kt = 0; kt < 3; ++kt) {
    const int key = m0 + kt * 16 + l15;
    v16h a0 = {}, a1 = {};
    if (key >= 0 && key < 512) {
      const _Float16* krow = klh + (b * 512 + key) * DMODEL + cbase;
      a0 = mk16h(krow + kbA,      krow + 16 + kbA);
      a1 = mk16h(krow + 32 + kbA, krow + 48 + kbA);
    }
#pragma unroll
    for (int jt = 0; jt < 2; ++jt) {
      v8f acc = {};
      acc = wmma32(a0, bq0[jt], acc);
      acc = wmma32(a1, bq1[jt], acc);
#pragma unroll
      for (int r = 0; r < 8; ++r) R[kt * 16 + r + hi * 8][jt * 16 + l15] = acc[r];
    }
  }

  // ---- stage V^T into LDS (rows kloc>=48 or invalid keys -> 0) ----
  for (int idx = lane; idx < 64 * 64; idx += 32) {
    const int dcol = idx & 63;
    const int kloc = idx >> 6;
    const int key  = m0 + kloc;
    _Float16 val = (_Float16)0.0f;
    if (kloc < 48 && key >= 0 && key < 512)
      val = vlh[(b * 512 + key) * DMODEL + cbase + dcol];
    VT[dcol][kloc] = val;
  }

  __syncthreads();

  // ---- banded softmax (one lane per query row) ----
  if (lane < 16) {
    const int nl = lane;
    float mx = -3.0e38f;
#pragma unroll 1
    for (int c = 0; c < 48; ++c) {
      const int m   = m0 + c;
      const int dlt = (n0 + nl) - m;          // = nl + 16 - c
      float s = -1.0e30f;
      if (m >= 0 && m < 512 && dlt >= -8 && dlt <= 8) {
        const int j = dlt + 8;                // 0..16
        s = (S[nl][c] + P[nl][j] + R[c][j]) * RSCALE;
        mx = fmaxf(mx, s);
      }
      S[nl][c] = s;
    }
    float sum = 0.0f;
#pragma unroll 1
    for (int c = 0; c < 48; ++c) {
      const float p = __expf(S[nl][c] - mx);  // masked entries underflow to 0
      S[nl][c] = p;
      sum += p;
    }
    const float rs = 1.0f / sum;
#pragma unroll 1
    for (int c = 0; c < 48; ++c) PR[nl][c] = (_Float16)(S[nl][c] * rs);
#pragma unroll 1
    for (int c = 48; c < 64; ++c) PR[nl][c] = (_Float16)0.0f;
  }

  __syncthreads();

  // ---- ctx = P (16x48, padded to 64) x Vwin (48x64) via WMMA ----
  const _Float16* prrow = &PR[l15][0];
  v16h ap0 = mk16h(prrow + kbA,      prrow + 16 + kbA);
  v16h ap1 = mk16h(prrow + 32 + kbA, prrow + 48 + kbA);

#pragma unroll
  for (int nt = 0; nt < 4; ++nt) {
    const _Float16* vtr = &VT[nt * 16 + l15][0];
    v16h b0 = mk16h(vtr + kbA,      vtr + 16 + kbA);
    v16h b1 = mk16h(vtr + 32 + kbA, vtr + 48 + kbA);
    v8f acc = {};
    acc = wmma32(ap0, b0, acc);
    acc = wmma32(ap1, b1, acc);
#pragma unroll
    for (int r = 0; r < 8; ++r) {
      const int n = n0 + r + hi * 8;
      out[(b * 512 + n) * DMODEL + cbase + nt * 16 + l15] = acc[r];
    }
  }
}

// ---- launcher ---------------------------------------------------------------
extern "C" void kernel_launch(void* const* d_in, const int* in_sizes, int n_in,
                              void* d_out, int out_size, void* d_ws, size_t ws_size,
                              hipStream_t stream) {
  (void)in_sizes; (void)n_in; (void)out_size; (void)ws_size;
  const float* q   = (const float*)d_in[0];
  const float* k   = (const float*)d_in[1];
  const float* v   = (const float*)d_in[2];
  const float* rel = (const float*)d_in[3];
  const float* Wq  = (const float*)d_in[4];
  const float* bq  = (const float*)d_in[5];
  const float* Wk  = (const float*)d_in[6];
  const float* bk  = (const float*)d_in[7];
  const float* Wv  = (const float*)d_in[8];
  const float* bv  = (const float*)d_in[9];

  char* ws = (char*)d_ws;
  _Float16* qlh   = (_Float16*)(ws);                      // 8192x512 f16 = 8 MiB
  _Float16* klh   = (_Float16*)(ws + 8388608);
  _Float16* vlh   = (_Float16*)(ws + 16777216);
  _Float16* qh    = (_Float16*)(ws + 25165824);           // f16 copies of inputs
  _Float16* kh    = (_Float16*)(ws + 33554432);
  _Float16* vh    = (_Float16*)(ws + 41943040);
  _Float16* Wqh   = (_Float16*)(ws + 50331648);           // 512x512 f16 = 512 KiB
  _Float16* Wkh   = (_Float16*)(ws + 50855936);
  _Float16* Wvh   = (_Float16*)(ws + 51380224);
  _Float16* relh  = (_Float16*)(ws + 51904512);           // rel rows 504..535, 32x512
  _Float16* poskw = (_Float16*)(ws + 51937280);
  _Float16* posqw = (_Float16*)(ws + 51970048);

  const int NX = 8192 * DMODEL;   // 4194304
  const int NW = DMODEL * DMODEL; // 262144
  const int NR = 32 * DMODEL;     // 16384

  // Stage 0: conversions
  cvt_f16_kernel<<<NX / (8 * 256), 256, 0, stream>>>(q, qh, NX);
  cvt_f16_kernel<<<NX / (8 * 256), 256, 0, stream>>>(k, kh, NX);
  cvt_f16_kernel<<<NX / (8 * 256), 256, 0, stream>>>(v, vh, NX);
  cvt_f16_kernel<<<NW / (8 * 256), 256, 0, stream>>>(Wq, Wqh, NW);
  cvt_f16_kernel<<<NW / (8 * 256), 256, 0, stream>>>(Wk, Wkh, NW);
  cvt_f16_kernel<<<NW / (8 * 256), 256, 0, stream>>>(Wv, Wvh, NW);
  cvt_f16_kernel<<<NR / (8 * 256), 256, 0, stream>>>(rel + 504 * DMODEL, relh, NR);

  // Stage 1: projections (32-row macro tiles)
  dim3 blk(32);
  gemm_xwT_kernel<<<dim3(256, 8), blk, 0, stream>>>(qh, Wqh, bq, qlh);
  gemm_xwT_kernel<<<dim3(256, 8), blk, 0, stream>>>(kh, Wkh, bk, klh);
  gemm_xwT_kernel<<<dim3(256, 8), blk, 0, stream>>>(vh, Wvh, bv, vlh);
  gemm_xwT_kernel<<<dim3(1, 8), blk, 0, stream>>>(relh, Wkh, bk, poskw);
  gemm_xwT_kernel<<<dim3(1, 8), blk, 0, stream>>>(relh, Wqh, bq, posqw);

  // Stage 2: banded attention
  attn_kernel<<<dim3(32, 8, 16), blk, 0, stream>>>(qlh, klh, vlh, poskw, posqw,
                                                   (float*)d_out);
}
